// DProjPythiaModel_10960756540197
// MI455X (gfx1250) — matrix operand order repaired and checked by reading, not compiled, not run on hardware
//
#include <hip/hip_runtime.h>
#include <hip/hip_bf16.h>
#include <cstdint>

// ---------------------------------------------------------------------------
// Model constants (from reference)
// ---------------------------------------------------------------------------
constexpr int Vv = 50304, Ee = 512, Pp = 320, NLn = 6, Hh = 8, Ii = 1024;
constexpr int Bb = 2, Ss = 1024;
constexpr int HDd = Pp / Hh;        // 40
constexpr int ROTr = 10;            // int(40*0.25)
constexpr int M_TOK = Bb * Ss;      // 2048 tokens

typedef __bf16 bf16;
typedef __attribute__((ext_vector_type(16))) __bf16 v16bf;
typedef __attribute__((ext_vector_type(8)))  float  v8f;
typedef int v4i __attribute__((vector_size(16)));   // matches builtin param type

#if defined(__HIP_DEVICE_COMPILE__) && __has_builtin(__builtin_amdgcn_global_load_async_to_lds_b128)
#define HAVE_ASYNC_LDS 1
#else
#define HAVE_ASYNC_LDS 0
#endif

#if HAVE_ASYNC_LDS
// 16-byte async DMA: global (AS1) -> LDS (AS3). Generic->AS casts are
// addrspacecasts (correct for LDS; no inttoptr).
#define ASYNC_CP16(dstLds, srcGlb)                                            \
  __builtin_amdgcn_global_load_async_to_lds_b128(                             \
      (__attribute__((address_space(1))) v4i*)(v4i*)(srcGlb),                 \
      (__attribute__((address_space(3))) v4i*)(v4i*)(dstLds), 0, 0)
#if __has_builtin(__builtin_amdgcn_s_wait_asynccnt)
#define WAIT_ASYNC(n) __builtin_amdgcn_s_wait_asynccnt(n)
#else
#define WAIT_ASYNC(n) asm volatile("s_wait_asynccnt %0" ::"i"(n) : "memory")
#endif
#endif

// ---------------------------------------------------------------------------
// Embedding gather
// ---------------------------------------------------------------------------
__global__ void gather_kernel(const int* __restrict__ ids,
                              const float* __restrict__ embed_w,
                              float* __restrict__ out) {
  size_t i = (size_t)blockIdx.x * blockDim.x + threadIdx.x;
  size_t total = (size_t)M_TOK * Ee;
  if (i >= total) return;
  int tok = (int)(i / Ee);
  int e   = (int)(i % Ee);
  out[i] = embed_w[(size_t)ids[tok] * Ee + e];
}

// ---------------------------------------------------------------------------
// Generic LayerNorm over last dim D (one block per row, 256 threads)
// ---------------------------------------------------------------------------
__global__ void ln_kernel(const float* __restrict__ X,
                          const float* __restrict__ w,
                          const float* __restrict__ b,
                          float* __restrict__ Y, int D) {
  __shared__ float red[64];
  const int row = blockIdx.x;
  const float* x = X + (size_t)row * D;
  float s = 0.f, ss = 0.f;
  for (int i = threadIdx.x; i < D; i += blockDim.x) {
    float v = x[i]; s += v; ss += v * v;
  }
  for (int off = 16; off; off >>= 1) {
    s  += __shfl_down(s,  off, 32);
    ss += __shfl_down(ss, off, 32);
  }
  int wid = threadIdx.x >> 5;
  if ((threadIdx.x & 31) == 0) { red[wid] = s; red[32 + wid] = ss; }
  __syncthreads();
  if (threadIdx.x == 0) {
    float ts = 0.f, tss = 0.f;
    int nw = blockDim.x >> 5;
    for (int i = 0; i < nw; ++i) { ts += red[i]; tss += red[32 + i]; }
    float m = ts / D;
    red[0] = m;
    red[1] = tss / D - m * m;
  }
  __syncthreads();
  float m = red[0];
  float inv = rsqrtf(red[1] + 1e-5f);
  float* y = Y + (size_t)row * D;
  for (int i = threadIdx.x; i < D; i += blockDim.x)
    y[i] = (x[i] - m) * inv * w[i] + b[i];
}

// ---------------------------------------------------------------------------
// Sequential dproj scan: h_t = tanh(LN([h_{t-1}, te_t] @ W + b))
// ---------------------------------------------------------------------------
__global__ void dproj_scan_kernel(const float* __restrict__ te,
                                  const float* __restrict__ W,   // (832,320)
                                  const float* __restrict__ bias,
                                  const float* __restrict__ lnw,
                                  const float* __restrict__ lnb,
                                  float* __restrict__ X) {       // (B,S,P)
  __shared__ float hprev[2][Pp];
  __shared__ float wsum[2][16];
  __shared__ float stats[4];
  const int j = threadIdx.x;          // 0..319
  const int lane = j & 31, wid = j >> 5;
  hprev[0][j] = 0.f; hprev[1][j] = 0.f;
  __syncthreads();

  for (int t = 0; t < Ss; ++t) {
    float a0 = bias[j], a1 = bias[j];
    const float* wc = W + j;                       // column j, stride 320
    #pragma unroll 4
    for (int i = 0; i < Pp; ++i) {
      float wv = wc[(size_t)i * Pp];
      a0 += hprev[0][i] * wv;
      a1 += hprev[1][i] * wv;
    }
    const float* e0 = te + (size_t)(0 * Ss + t) * Ee;
    const float* e1 = te + (size_t)(1 * Ss + t) * Ee;
    const float* w2 = W + (size_t)Pp * Pp + j;
    #pragma unroll 4
    for (int i = 0; i < Ee; ++i) {
      float wv = w2[(size_t)i * Pp];
      a0 += e0[i] * wv;
      a1 += e1[i] * wv;
    }
    // mean
    float s0 = a0, s1 = a1;
    for (int off = 16; off; off >>= 1) {
      s0 += __shfl_down(s0, off, 32);
      s1 += __shfl_down(s1, off, 32);
    }
    if (lane == 0) { wsum[0][wid] = s0; wsum[1][wid] = s1; }
    __syncthreads();
    if (j == 0) {
      float m0 = 0.f, m1 = 0.f;
      for (int i = 0; i < 10; ++i) { m0 += wsum[0][i]; m1 += wsum[1][i]; }
      stats[0] = m0 / (float)Pp; stats[1] = m1 / (float)Pp;
    }
    __syncthreads();
    float d0 = a0 - stats[0], d1 = a1 - stats[1];
    // variance
    s0 = d0 * d0; s1 = d1 * d1;
    for (int off = 16; off; off >>= 1) {
      s0 += __shfl_down(s0, off, 32);
      s1 += __shfl_down(s1, off, 32);
    }
    if (lane == 0) { wsum[0][wid] = s0; wsum[1][wid] = s1; }
    __syncthreads();
    if (j == 0) {
      float v0 = 0.f, v1 = 0.f;
      for (int i = 0; i < 10; ++i) { v0 += wsum[0][i]; v1 += wsum[1][i]; }
      stats[2] = rsqrtf(v0 / (float)Pp + 1e-5f);
      stats[3] = rsqrtf(v1 / (float)Pp + 1e-5f);
    }
    __syncthreads();
    float h0 = tanhf(d0 * stats[2] * lnw[j] + lnb[j]);
    float h1 = tanhf(d1 * stats[3] * lnw[j] + lnb[j]);
    hprev[0][j] = h0; hprev[1][j] = h1;
    X[(size_t)(0 * Ss + t) * Pp + j] = h0;
    X[(size_t)(1 * Ss + t) * Pp + j] = h1;
    __syncthreads();
  }
}

// ---------------------------------------------------------------------------
// Pack B (K x N, or transposed source N x K) into WMMA bf16 per-lane layout:
//   out[nTile][kTile][lane][e], lane = hi*16 + (n&15);
//   e<8 -> k = hi*8 + e ; e>=8 -> k = 16 + hi*8 + (e-8)
// ---------------------------------------------------------------------------
__global__ void pack_b_kernel(const float* __restrict__ W,
                              bf16* __restrict__ out,
                              int K, int N, int transposed) {
  size_t i = (size_t)blockIdx.x * blockDim.x + threadIdx.x;
  size_t total = (size_t)K * N;
  if (i >= total) return;
  int k, n;
  if (transposed) { n = (int)(i / K); k = (int)(i - (size_t)n * K); }
  else            { k = (int)(i / N); n = (int)(i - (size_t)k * N); }
  float val = W[i];
  int kt = k >> 5, kin = k & 31;
  int hi = (kin >> 3) & 1, e = (kin & 7) + ((kin >> 4) << 3);
  int lane = hi * 16 + (n & 15);
  size_t dst = ((((size_t)(n >> 4)) * (size_t)(K >> 5) + kt) * 32 + lane) * 16 + e;
  out[dst] = (bf16)val;
}

// ---------------------------------------------------------------------------
// Pack A (M x K f32 row-major) into WMMA bf16 per-lane A layout:
//   out[mTile][kTile][lane][e], lane = hi*16 + (m&15);
//   e<8 -> k = hi*8 + e ; e>=8 -> k = 16 + hi*8 + (e-8)
// ---------------------------------------------------------------------------
__global__ void pack_a_kernel(const float* __restrict__ A,
                              bf16* __restrict__ out,
                              int M, int K) {
  size_t i = (size_t)blockIdx.x * blockDim.x + threadIdx.x;
  size_t total = (size_t)M * K;
  if (i >= total) return;
  int m = (int)(i / K);
  int k = (int)(i - (size_t)m * K);
  int kt = k >> 5, kin = k & 31;
  int hi = (kin >> 3) & 1;
  int e  = (kin & 7) + ((kin >> 4) << 3);
  int lane = hi * 16 + (m & 15);
  size_t dst = ((((size_t)(m >> 4)) * (size_t)(K >> 5) + kt) * 32 + lane) * 16 + e;
  out[dst] = (bf16)A[i];
}

// ---------------------------------------------------------------------------
// WMMA GEMM: C[M,N] = act(A @ B + bias); A and B both pre-packed bf16,
// f32 accumulate. One wave per block; each wave owns a 32(M) x 64(N) tile:
// 2 A-tiles x 4 B-tiles = 8 WMMAs per K-step of 32.
// A panels are streamed through LDS with the async-DMA path when available.
// act: 0 = none, 1 = gelu(tanh approx)
// ---------------------------------------------------------------------------
__device__ __forceinline__ float gelu_tanh(float v) {
  return 0.5f * v * (1.0f + tanhf(0.7978845608028654f * (v + 0.044715f * v * v * v)));
}

__global__ void gemm_bf16_wmma(const bf16* __restrict__ Ap,
                               const bf16* __restrict__ Bp,
                               const float* __restrict__ bias,
                               float* __restrict__ C,
                               int M, int N, int K, int act) {
  const int lane = threadIdx.x & 31;
  const int lo = lane & 15, hi = lane >> 4;
  const int n0 = blockIdx.x * 64;
  const int m0 = blockIdx.y * 32;
  const int kt = K >> 5;
  const size_t tileStride = (size_t)kt * 512;   // bf16 elems per 16-row/col tile

  v8f acc[2][4];
  #pragma unroll
  for (int mt = 0; mt < 2; ++mt)
    #pragma unroll
    for (int t = 0; t < 4; ++t)
      acc[mt][t] = (v8f){0.f,0.f,0.f,0.f,0.f,0.f,0.f,0.f};

  const bf16* aBase0 = Ap + (size_t)(m0 >> 4) * tileStride + (size_t)lane * 16;
  const bf16* aBase1 = aBase0 + tileStride;

#if HAVE_ASYNC_LDS
  __shared__ alignas(32) bf16 aSh[2][1024];     // double-buffered 32x32 A panel
  {
    bf16* d0 = &aSh[0][(size_t)lane * 16];
    bf16* d1 = &aSh[0][512 + (size_t)lane * 16];
    ASYNC_CP16(d0,     aBase0);
    ASYNC_CP16(d0 + 8, aBase0 + 8);
    ASYNC_CP16(d1,     aBase1);
    ASYNC_CP16(d1 + 8, aBase1 + 8);
  }
#endif

  for (int kk = 0; kk < K; kk += 32) {
    const int ktile = kk >> 5;
#if HAVE_ASYNC_LDS
    const int buf = ktile & 1;
    if (kk + 32 < K) {
      const bf16* s0 = aBase0 + (size_t)(ktile + 1) * 512;
      const bf16* s1 = aBase1 + (size_t)(ktile + 1) * 512;
      bf16* d0 = &aSh[buf ^ 1][(size_t)lane * 16];
      bf16* d1 = &aSh[buf ^ 1][512 + (size_t)lane * 16];
      ASYNC_CP16(d0,     s0);
      ASYNC_CP16(d0 + 8, s0 + 8);
      ASYNC_CP16(d1,     s1);
      ASYNC_CP16(d1 + 8, s1 + 8);
      WAIT_ASYNC(4);    // current panel's 4 transfers done (in-order)
    } else {
      WAIT_ASYNC(0);
    }
    v16bf a0 = *reinterpret_cast<const v16bf*>(&aSh[buf][(size_t)lane * 16]);
    v16bf a1 = *reinterpret_cast<const v16bf*>(&aSh[buf][512 + (size_t)lane * 16]);
#else
    v16bf a0 = *reinterpret_cast<const v16bf*>(aBase0 + (size_t)ktile * 512);
    v16bf a1 = *reinterpret_cast<const v16bf*>(aBase1 + (size_t)ktile * 512);
#endif

    const bf16* bbase = Bp + ((size_t)(n0 >> 4) * kt + ktile) * 512 + (size_t)lane * 16;
    if (kk + 32 < K) __builtin_prefetch(bbase + 512, 0, 3);
    #pragma unroll
    for (int t = 0; t < 4; ++t) {
      v16bf b = *reinterpret_cast<const v16bf*>(bbase + (size_t)t * tileStride);
      acc[0][t] = __builtin_amdgcn_wmma_f32_16x16x32_bf16(
          false, a0, false, b, (short)0, acc[0][t], false, false);
      acc[1][t] = __builtin_amdgcn_wmma_f32_16x16x32_bf16(
          false, a1, false, b, (short)0, acc[1][t], false, false);
    }
  }

  #pragma unroll
  for (int mt = 0; mt < 2; ++mt) {
    #pragma unroll
    for (int t = 0; t < 4; ++t) {
      int col = n0 + t * 16 + lo;
      float bv = bias ? bias[col] : 0.f;
      #pragma unroll
      for (int r = 0; r < 8; ++r) {
        int row = m0 + mt * 16 + hi * 8 + r;
        float v = acc[mt][t][r] + bv;
        if (act == 1) v = gelu_tanh(v);
        C[(size_t)row * N + col] = v;
      }
    }
  }
}

// ---------------------------------------------------------------------------
// RoPE in place on qkv buffer, layout (b,s,H,3*HD). One thread per (b,s,h).
// ---------------------------------------------------------------------------
__global__ void rope_kernel(float* __restrict__ qkv) {
  int idx = blockIdx.x * blockDim.x + threadIdx.x;   // over B*S*H
  if (idx >= Bb * Ss * Hh) return;
  int t = (idx / Hh) % Ss;
  float* base = qkv + (size_t)idx * (3 * HDd);
  float c[ROTr], sn[ROTr];
  #pragma unroll
  for (int d = 0; d < ROTr; ++d) {
    float inv = powf(10000.0f, -(2.0f * (d % 5)) / (float)ROTr);
    float f = (float)t * inv;
    c[d] = cosf(f); sn[d] = sinf(f);
  }
  #pragma unroll
  for (int part = 0; part < 2; ++part) {   // q then k
    float* p = base + part * HDd;
    float tv[ROTr];
    #pragma unroll
    for (int d = 0; d < ROTr; ++d) tv[d] = p[d];
    #pragma unroll
    for (int d = 0; d < ROTr; ++d) {
      float rot = (d < 5) ? -tv[d + 5] : tv[d - 5];
      p[d] = tv[d] * c[d] + rot * sn[d];
    }
  }
}

// ---------------------------------------------------------------------------
// Fused causal attention (streaming, no SxS matrix): one block per (b,h,q).
// ---------------------------------------------------------------------------
__global__ void attn_kernel(const float* __restrict__ qkv,
                            float* __restrict__ ctx) {
  __shared__ float sc[Ss];
  __shared__ float qv[HDd];
  __shared__ float red[128];
  const int qpos = blockIdx.x, h = blockIdx.y, b = blockIdx.z;
  const int tid = threadIdx.x;  // 128 threads
  const float* qbase = qkv + (((size_t)(b * Ss + qpos) * Hh + h) * (3 * HDd));
  if (tid < HDd) qv[tid] = qbase[tid];
  __syncthreads();

  const float scale = 0.15811388300841897f;  // 1/sqrt(40)
  const int nk = qpos + 1;
  for (int t = tid; t < nk; t += 128) {
    const float* kp = qkv + (((size_t)(b * Ss + t) * Hh + h) * (3 * HDd)) + HDd;
    float d = 0.f;
    #pragma unroll
    for (int e = 0; e < HDd; ++e) d += qv[e] * kp[e];
    sc[t] = d * scale;
  }
  __syncthreads();

  float mx = -1e30f;
  for (int t = tid; t < nk; t += 128) mx = fmaxf(mx, sc[t]);
  red[tid] = mx; __syncthreads();
  for (int off = 64; off; off >>= 1) {
    if (tid < off) red[tid] = fmaxf(red[tid], red[tid + off]);
    __syncthreads();
  }
  mx = red[0];

  float sum = 0.f;
  for (int t = tid; t < nk; t += 128) {
    float e = expf(sc[t] - mx);
    sc[t] = e; sum += e;
  }
  __syncthreads();
  red[tid] = sum; __syncthreads();
  for (int off = 64; off; off >>= 1) {
    if (tid < off) red[tid] += red[tid + off];
    __syncthreads();
  }
  float rs = 1.0f / red[0];

  if (tid < HDd) {
    const float* vp = qkv + (((size_t)(b * Ss) * Hh + h) * (3 * HDd)) + 2 * HDd + tid;
    float acc = 0.f;
    for (int t = 0; t < nk; ++t) acc += sc[t] * vp[(size_t)t * (Hh * 3 * HDd)];
    ctx[(((size_t)(b * Ss + qpos)) * Hh + h) * HDd + tid] = acc * rs;
  }
}

// ---------------------------------------------------------------------------
// x += a + m
// ---------------------------------------------------------------------------
__global__ void add3_kernel(float* __restrict__ x,
                            const float* __restrict__ a,
                            const float* __restrict__ m, size_t n) {
  size_t i = (size_t)blockIdx.x * blockDim.x + threadIdx.x;
  if (i < n) x[i] += a[i] + m[i];
}

// ---------------------------------------------------------------------------
// Host side
// ---------------------------------------------------------------------------
extern "C" void kernel_launch(void* const* d_in, const int* in_sizes, int n_in,
                              void* d_out, int out_size, void* d_ws, size_t ws_size,
                              hipStream_t stream) {
  (void)in_sizes; (void)n_in; (void)out_size; (void)ws_size;

  const int*   input_ids    = (const int*)  d_in[0];
  const float* embed_w      = (const float*)d_in[1];
  const float* embed_norm_w = (const float*)d_in[2];
  const float* embed_norm_b = (const float*)d_in[3];
  const float* dproj_w      = (const float*)d_in[4];
  const float* dproj_b      = (const float*)d_in[5];
  const float* dproj_ln_w   = (const float*)d_in[6];
  const float* dproj_ln_b   = (const float*)d_in[7];
  const float* ln1_w        = (const float*)d_in[8];
  const float* ln1_b        = (const float*)d_in[9];
  const float* ln2_w        = (const float*)d_in[10];
  const float* ln2_b        = (const float*)d_in[11];
  const float* qkv_w        = (const float*)d_in[12];
  const float* qkv_b        = (const float*)d_in[13];
  const float* dense_w      = (const float*)d_in[14];
  const float* dense_b      = (const float*)d_in[15];
  const float* fc1_w        = (const float*)d_in[16];
  const float* fc1_b        = (const float*)d_in[17];
  const float* fc2_w        = (const float*)d_in[18];
  const float* fc2_b        = (const float*)d_in[19];
  const float* final_ln_w   = (const float*)d_in[20];
  const float* final_ln_b   = (const float*)d_in[21];
  const float* lm_head_w    = (const float*)d_in[22];

  // workspace carving (bytes, 256-aligned)
  char* ws = (char*)d_ws;
  size_t off = 0;
  auto carve = [&](size_t bytes) {
    char* p = ws + off;
    off += (bytes + 255) & ~(size_t)255;
    return p;
  };
  float* te     = (float*)carve((size_t)M_TOK * Ee * 4);       // 4.19 MB
  float* x      = (float*)carve((size_t)M_TOK * Pp * 4);       // 2.62 MB
  float* lnbuf  = (float*)carve((size_t)M_TOK * Pp * 4);
  float* qkvb   = (float*)carve((size_t)M_TOK * 3 * Pp * 4);   // 7.86 MB
  float* ctx    = (float*)carve((size_t)M_TOK * Pp * 4);
  float* aout   = (float*)carve((size_t)M_TOK * Pp * 4);
  float* m1     = (float*)carve((size_t)M_TOK * Ii * 4);       // 8.39 MB
  float* mout   = (float*)carve((size_t)M_TOK * Pp * 4);
  bf16*  packB  = (bf16*) carve((size_t)Vv * Pp * 2);          // 32.2 MB (largest weight)
  bf16*  packA  = (bf16*) carve((size_t)M_TOK * Ii * 2);       // 4.19 MB (largest A)

  auto packA_launch = [&](const float* A, int M, int K) {
    size_t total = (size_t)M * K;
    pack_a_kernel<<<(unsigned)((total + 255) / 256), 256, 0, stream>>>(A, packA, M, K);
  };
  auto packB_launch = [&](const float* W, int K, int N, int transposed) {
    size_t total = (size_t)K * N;
    pack_b_kernel<<<(unsigned)((total + 255) / 256), 256, 0, stream>>>(W, packB, K, N, transposed);
  };
  auto gemm_launch = [&](const float* bias, float* C, int M, int N, int K, int act) {
    gemm_bf16_wmma<<<dim3(N / 64, M / 32), 32, 0, stream>>>(packA, packB, bias, C, M, N, K, act);
  };

  // 1) embedding gather + LN
  {
    size_t total = (size_t)M_TOK * Ee;
    gather_kernel<<<(unsigned)((total + 255) / 256), 256, 0, stream>>>(input_ids, embed_w, te);
    ln_kernel<<<M_TOK, 256, 0, stream>>>(te, embed_norm_w, embed_norm_b, te, Ee);
  }

  // 2) sequential dproj scan
  dproj_scan_kernel<<<1, Pp, 0, stream>>>(te, dproj_w, dproj_b, dproj_ln_w, dproj_ln_b, x);

  const size_t xElems = (size_t)M_TOK * Pp;

  // 3) transformer layers
  for (int l = 0; l < NLn; ++l) {
    const float* qkvW = qkv_w   + (size_t)l * Pp * (3 * Pp);
    const float* qkvB = qkv_b   + (size_t)l * (3 * Pp);
    const float* dW   = dense_w + (size_t)l * Pp * Pp;
    const float* dB   = dense_b + (size_t)l * Pp;
    const float* f1W  = fc1_w   + (size_t)l * Pp * Ii;
    const float* f1B  = fc1_b   + (size_t)l * Ii;
    const float* f2W  = fc2_w   + (size_t)l * Ii * Pp;
    const float* f2B  = fc2_b   + (size_t)l * Pp;

    // attention branch
    ln_kernel<<<M_TOK, 256, 0, stream>>>(x, ln1_w + (size_t)l * Pp, ln1_b + (size_t)l * Pp, lnbuf, Pp);
    packA_launch(lnbuf, M_TOK, Pp);
    packB_launch(qkvW, Pp, 3 * Pp, 0);
    gemm_launch(qkvB, qkvb, M_TOK, 3 * Pp, Pp, 0);
    rope_kernel<<<(Bb * Ss * Hh + 255) / 256, 256, 0, stream>>>(qkvb);
    attn_kernel<<<dim3(Ss, Hh, Bb), 128, 0, stream>>>(qkvb, ctx);
    packA_launch(ctx, M_TOK, Pp);
    packB_launch(dW, Pp, Pp, 0);
    gemm_launch(dB, aout, M_TOK, Pp, Pp, 0);

    // MLP branch (reads pre-residual x)
    ln_kernel<<<M_TOK, 256, 0, stream>>>(x, ln2_w + (size_t)l * Pp, ln2_b + (size_t)l * Pp, lnbuf, Pp);
    packA_launch(lnbuf, M_TOK, Pp);
    packB_launch(f1W, Pp, Ii, 0);
    gemm_launch(f1B, m1, M_TOK, Ii, Pp, 1);
    packA_launch(m1, M_TOK, Ii);
    packB_launch(f2W, Ii, Pp, 0);
    gemm_launch(f2B, mout, M_TOK, Pp, Ii, 0);

    add3_kernel<<<(unsigned)((xElems + 255) / 256), 256, 0, stream>>>(x, aout, mout, xElems);
  }

  // 4) final LN + lm_head (weight is (V,P): transposed pack so col v = row of W)
  ln_kernel<<<M_TOK, 256, 0, stream>>>(x, final_ln_w, final_ln_b, lnbuf, Pp);
  packA_launch(lnbuf, M_TOK, Pp);
  packB_launch(lm_head_w, Pp, Vv, 1);
  gemm_launch(nullptr, (float*)d_out, M_TOK, Vv, Pp, 0);
}
